// SwinBlock1D_37744172597867
// MI455X (gfx1250) — matrix operand order repaired and checked by reading, not compiled, not run on hardware
//
#include <hip/hip_runtime.h>
#include <hip/hip_bf16.h>

// ---------------------------------------------------------------------------
// SwinBlock1D for MI455X (gfx1250): WMMA GEMMs + Tensor Data Mover staging.
// All matmuls: v_wmma_f32_16x16x32_f16 (f16 A/B, fp32 accumulate).
// GEMM slabs are DMA'd into LDS by the TDM (tensor_load_to_lds, TENSORcnt),
// double-buffered so compute waves only issue ds_load + wmma.
// ---------------------------------------------------------------------------

typedef __attribute__((ext_vector_type(16))) _Float16 v16h;
typedef __attribute__((ext_vector_type(8)))  float    v8f;
typedef __attribute__((ext_vector_type(4)))  unsigned int u32x4;
typedef __attribute__((ext_vector_type(8)))  int      i32x8;
typedef __attribute__((ext_vector_type(4)))  int      i32x4;

struct alignas(16) H8 { _Float16 h[8]; };

constexpr int Bc    = 16;
constexpr int Cc    = 256;
constexpr int Tc    = 8192;
constexpr int Hh    = 8;       // heads
constexpr int Wn    = 128;     // window
constexpr int NW    = Tc / Wn; // 64 windows
constexpr int SHIFT = Wn / 2;  // 64
constexpr int HID   = 4 * Cc;  // 1024
constexpr int MTOK  = Bc * Tc; // 131072 token rows

#define DEV static __device__ __forceinline__

// ---------------------------------------------------------------------------
// TDM: DMA a 2D tile (rows x 32 halves, row stride = stride_elems halves)
// from global memory into LDS at byte offset lds_addr (contiguous row-major
// [row][0..31]).  D# packing per CDNA5 ISA 8.3/8.4.  Issue from ONE wave.
// ---------------------------------------------------------------------------
DEV void tdm_load_2d(unsigned lds_addr, const _Float16* gptr,
                     unsigned stride_elems, unsigned rows, unsigned cols) {
  const unsigned long long ga = (unsigned long long)(uintptr_t)gptr;
  u32x4 g0;
  g0[0] = 1u;                                            // count=1 (valid user D#)
  g0[1] = lds_addr;                                      // lds_addr [63:32]
  g0[2] = (unsigned)(ga & 0xFFFFFFFFu);                  // global_addr lo
  g0[3] = (unsigned)((ga >> 32) & 0x1FFFFFFu) | (2u << 30); // addr hi | type=2

  const unsigned t0 = stride_elems;   // tensor_dim0 (row length, OOB bound)
  const unsigned t1 = rows;           // tensor_dim1
  const unsigned long long s0 = stride_elems; // tensor_dim0_stride
  i32x8 g1;
  g1[0] = (int)(1u << 16);            // wg_mask=0 | data_size=1 (2 bytes)
  g1[1] = (int)((t0 & 0xFFFFu) << 16);                  // tensor_dim0[15:0]
  g1[2] = (int)(((t0 >> 16) & 0xFFFFu) | ((t1 & 0xFFFFu) << 16));
  g1[3] = (int)(((t1 >> 16) & 0xFFFFu) | ((cols & 0xFFFFu) << 16)); // tile_dim0
  g1[4] = (int)(rows & 0xFFFFu);      // tile_dim1 | tile_dim2=0
  g1[5] = (int)(s0 & 0xFFFFFFFFu);    // tensor_dim0_stride lo
  g1[6] = (int)((s0 >> 32) & 0xFFFFu);// stride hi | tensor_dim1_stride lo = 0
  g1[7] = 0;
  const i32x4 gz = {0, 0, 0, 0};      // groups 2/3: 2D tensor
#if defined(__clang_major__) && (__clang_major__ >= 23)
  const i32x8 gz8 = {0, 0, 0, 0, 0, 0, 0, 0};
  __builtin_amdgcn_tensor_load_to_lds(g0, g1, gz, gz, gz8, 0);
#else
  __builtin_amdgcn_tensor_load_to_lds(g0, g1, gz, gz, 0);
#endif
}

// A-operand (16x32 f16, MxK): lane M = lane%16; element e<8 -> K = 8*half+e,
// element e>=8 -> K = 16 + 8*half + (e-8).  `row` points at A[M][k_base].
DEV v16h frag_a(const _Float16* row) {
  const int hf = (threadIdx.x >> 4) & 1;
  H8 lo = *reinterpret_cast<const H8*>(row + 8 * hf);
  H8 hi = *reinterpret_cast<const H8*>(row + 16 + 8 * hf);
  v16h a;
#pragma unroll
  for (int e = 0; e < 8; ++e) { a[e] = lo.h[e]; a[8 + e] = hi.h[e]; }
  return a;
}

// B-operand (32x16 f16, KxN) from N-major storage Bt[n][k]:
// lane N = lane%16; element e<8 -> K = 8*half+e, element e>=8 -> K = 16 + 8*half + (e-8).  `rowN` points at Bt[N][k_base].
DEV v16h frag_bT(const _Float16* rowN) {
  const int hf = (threadIdx.x >> 4) & 1;
  H8 lo = *reinterpret_cast<const H8*>(rowN + 8 * hf);
  H8 hi = *reinterpret_cast<const H8*>(rowN + 16 + 8 * hf);
  v16h b;
#pragma unroll
  for (int e = 0; e < 8; ++e) { b[e] = lo.h[e]; b[8 + e] = hi.h[e]; }
  return b;
}

DEV v8f wmma_f16(v16h a, v16h b, v8f c) {
  return __builtin_amdgcn_wmma_f32_16x16x32_f16(false, a, false, b, (short)0, c,
                                                false, false);
}

// ---------------------------------------------------------------------------
// Weight convert: fp32 -> f16, SAME layout (reference weights are (out,in) =
// (N,K) row-major, i.e. already the N-major layout the GEMM B tiles want).
// ---------------------------------------------------------------------------
__global__ void wcvt_kernel(const float* __restrict__ W, _Float16* __restrict__ Wh,
                            int n) {
  int idx = blockIdx.x * 256 + threadIdx.x;
  if (idx < n) Wh[idx] = (_Float16)W[idx];
}

// ---------------------------------------------------------------------------
// LN1 + token-shift(-64) + transpose (B,C,T)->(B,T,C), f16 out.
// ---------------------------------------------------------------------------
__global__ __launch_bounds__(256) void ln1_shift_kernel(
    const float* __restrict__ x, const float* __restrict__ g,
    const float* __restrict__ be, _Float16* __restrict__ hbuf) {
  __shared__ float tile[Cc * 16];  // tile[c*16 + j]
  const int blk = blockIdx.x;      // MTOK/16 blocks
  const int b  = blk / (Tc / 16);
  const int t0 = (blk % (Tc / 16)) * 16;
  const int tid = threadIdx.x;
  {
    const int jt = tid & 15, cb = tid >> 4;
#pragma unroll
    for (int cc = 0; cc < 16; ++cc) {
      const int c = cc * 16 + cb;
      tile[c * 16 + jt] = x[((size_t)b * Cc + c) * Tc + t0 + jt];
    }
  }
  __syncthreads();
  const int j = tid >> 4, idx = tid & 15;  // 16 threads per token (16-aligned)
  float s = 0.f;
#pragma unroll
  for (int kk = 0; kk < 16; ++kk) s += tile[(idx * 16 + kk) * 16 + j];
#pragma unroll
  for (int m = 1; m < 16; m <<= 1) s += __shfl_xor(s, m, 32);
  const float mu = s * (1.f / Cc);
  float v2 = 0.f;
#pragma unroll
  for (int kk = 0; kk < 16; ++kk) {
    float d = tile[(idx * 16 + kk) * 16 + j] - mu;
    v2 += d * d;
  }
#pragma unroll
  for (int m = 1; m < 16; m <<= 1) v2 += __shfl_xor(v2, m, 32);
  const float rstd = rsqrtf(v2 * (1.f / Cc) + 1e-5f);
  const int t = t0 + j;
  int ts = t - SHIFT; if (ts < 0) ts += Tc;  // h_shifted[ts] = h[ts+SHIFT]
  _Float16* dst = &hbuf[((size_t)b * Tc + ts) * Cc];
#pragma unroll
  for (int kk = 0; kk < 16; ++kk) {
    const int c = idx * 16 + kk;
    dst[c] = (_Float16)((tile[c * 16 + j] - mu) * rstd * g[c] + be[c]);
  }
}

// ---------------------------------------------------------------------------
// LN2 over x1 (B*T, C) rows -> f16.  One block per token.
// ---------------------------------------------------------------------------
__global__ __launch_bounds__(256) void ln2_kernel(
    const float* __restrict__ x1, const float* __restrict__ g,
    const float* __restrict__ be, _Float16* __restrict__ h2) {
  const size_t row = blockIdx.x;
  const int tid = threadIdx.x, lane = tid & 31, wv = tid >> 5;
  __shared__ float part[8];
  const float v = x1[row * Cc + tid];
  float s = v;
#pragma unroll
  for (int m = 1; m < 32; m <<= 1) s += __shfl_xor(s, m, 32);
  if (lane == 0) part[wv] = s;
  __syncthreads();
  float tot = 0.f;
#pragma unroll
  for (int i = 0; i < 8; ++i) tot += part[i];
  const float mu = tot * (1.f / Cc);
  __syncthreads();
  const float d = v - mu;
  float q = d * d;
#pragma unroll
  for (int m = 1; m < 32; m <<= 1) q += __shfl_xor(q, m, 32);
  if (lane == 0) part[wv] = q;
  __syncthreads();
  tot = 0.f;
#pragma unroll
  for (int i = 0; i < 8; ++i) tot += part[i];
  const float rstd = rsqrtf(tot * (1.f / Cc) + 1e-5f);
  h2[row * Cc + tid] = (_Float16)(d * rstd * g[tid] + be[tid]);
}

// ---------------------------------------------------------------------------
// 128x128-tile WMMA GEMM, TDM double-buffered.
//   C(M,N) = A(M,K) * Bw(N,K)^T  (Bw is N-major: original weight layout).
// Dynamic LDS 32KB: A0 @0, A1 @8192, B0 @16384, B1 @24576 (each 128x32 f16).
// Wave 0 drives the TDM pipeline; 8 waves x (2x4) WMMA tiles each.
// ---------------------------------------------------------------------------
enum { EPI_QKV = 0, EPI_OUTPROJ = 1, EPI_GELU = 2, EPI_MLP2 = 3 };

template <int EPI>
__global__ __launch_bounds__(256) void gemm_wmma_kernel(
    const _Float16* __restrict__ A, const _Float16* __restrict__ Bw,
    const float* __restrict__ bias, int K, int N,
    _Float16* __restrict__ outH, float* __restrict__ outF,
    const float* __restrict__ res) {
  extern __shared__ char smem[];
  const int tid = threadIdx.x;
  const int m0 = blockIdx.x * 128;
  const int n0 = blockIdx.y * 128;
  const int wv = tid >> 5, lane = tid & 31, hf = lane >> 4, l16 = lane & 15;
  const int wm = (wv >> 1) * 32;  // wave row offset within 128
  const int wn = (wv & 1) * 64;   // wave col offset within 128

  v8f zero = {};
  v8f acc[2][4];
#pragma unroll
  for (int tm = 0; tm < 2; ++tm)
#pragma unroll
    for (int tn = 0; tn < 4; ++tn) acc[tm][tn] = zero;

  if (wv == 0) {  // prologue: slab 0 into buffer 0
    tdm_load_2d(0u,      &A[(size_t)m0 * K], (unsigned)K, 128, 32);
    tdm_load_2d(16384u,  &Bw[(size_t)n0 * K], (unsigned)K, 128, 32);
  }

  int cur = 0;
  for (int k0 = 0; k0 < K; k0 += 32) {
    if (wv == 0) {
      if (k0 + 32 < K) {  // issue next slab, then wait for current (2 newest)
        const unsigned nb = (unsigned)(cur ^ 1) * 8192u;
        tdm_load_2d(nb,          &A[(size_t)m0 * K + k0 + 32], (unsigned)K, 128, 32);
        tdm_load_2d(16384u + nb, &Bw[(size_t)n0 * K + k0 + 32], (unsigned)K, 128, 32);
        __builtin_amdgcn_s_wait_tensorcnt(2);
      } else {
        __builtin_amdgcn_s_wait_tensorcnt(0);
      }
    }
    __syncthreads();  // slab `cur` visible to all waves
    const _Float16* As = (const _Float16*)(smem + (size_t)cur * 8192);
    const _Float16* Bs = (const _Float16*)(smem + 16384 + (size_t)cur * 8192);
    v16h af[2], bf[4];
#pragma unroll
    for (int tm = 0; tm < 2; ++tm) af[tm] = frag_a(&As[(wm + tm * 16 + l16) * 32]);
#pragma unroll
    for (int tn = 0; tn < 4; ++tn) bf[tn] = frag_bT(&Bs[(wn + tn * 16 + l16) * 32]);
#pragma unroll
    for (int tm = 0; tm < 2; ++tm)
#pragma unroll
      for (int tn = 0; tn < 4; ++tn)
        acc[tm][tn] = wmma_f16(af[tm], bf[tn], acc[tm][tn]);
    __syncthreads();  // all reads of `cur` done before TDM overwrites it
    cur ^= 1;
  }

  // Epilogue. D layout: VGPR rr -> row rr + 8*half, col = lane%16.
#pragma unroll
  for (int tm = 0; tm < 2; ++tm) {
#pragma unroll
    for (int tn = 0; tn < 4; ++tn) {
#pragma unroll
      for (int rr = 0; rr < 8; ++rr) {
        const int grow = m0 + wm + tm * 16 + 8 * hf + rr;
        const int gcol = n0 + wn + tn * 16 + l16;
        float v = acc[tm][tn][rr] + bias[gcol];
        if (EPI == EPI_QKV) {
          outH[(size_t)grow * N + gcol] = (_Float16)v;
        } else if (EPI == EPI_GELU) {
          v = 0.5f * v * (1.f + erff(v * 0.70710678118654752f));
          outH[(size_t)grow * N + gcol] = (_Float16)v;
        } else if (EPI == EPI_OUTPROJ) {
          // rows are in shifted token order; un-shift, add x (read transposed
          // from (B,C,T)), write x1 in natural (B,T,C) order.
          const int b = grow / Tc, ts = grow % Tc;
          int t = ts + SHIFT; if (t >= Tc) t -= Tc;
          v += res[((size_t)b * Cc + gcol) * Tc + t];
          outF[((size_t)b * Tc + t) * Cc + gcol] = v;
        } else {  // EPI_MLP2: + x1 residual, store transposed to (B,C,T)
          v += res[(size_t)grow * Cc + gcol];
          const int b = grow / Tc, t = grow % Tc;
          outF[((size_t)b * Cc + gcol) * Tc + t] = v;
        }
      }
    }
  }
}

// ---------------------------------------------------------------------------
// Windowed attention: one block per (batch, window, head).
// Q/K head tiles (128x32, row stride 768) DMA'd by TDM; V staged transposed
// by the threads (TDM cannot transpose).  S = QK^T (1 WMMA per 16x16 tile,
// K = dh = 32), in-register softmax (16-lane shfl_xor), P -> LDS f16,
// O = P V (2 col-tiles x 4 K-steps per wave).
// Dynamic LDS 56KB: Qs @0, Ks @8192, Vt @16384, Ps @24576.
// ---------------------------------------------------------------------------
__global__ __launch_bounds__(256) void attn_kernel(
    const _Float16* __restrict__ qkv, _Float16* __restrict__ obuf) {
  extern __shared__ char smem[];
  _Float16* Qs  = (_Float16*)smem;            // [q][d]  128x32
  _Float16* Ks  = (_Float16*)(smem + 8192);   // [k][d]  128x32
  _Float16* Vts = (_Float16*)(smem + 16384);  // [d][k]  32x128 (transposed)
  _Float16* Ps  = (_Float16*)(smem + 24576);  // [q][k]  128x128
  const int tid = threadIdx.x;
  const int blk = blockIdx.x;
  const int b = blk / (NW * Hh);
  const int w = (blk / Hh) % NW;
  const int h = blk % Hh;
  const size_t rowbase = (size_t)b * Tc + (size_t)w * Wn;
  const int wv = tid >> 5, lane = tid & 31, hf = lane >> 4, l16 = lane & 15;

  const _Float16* base = qkv + rowbase * (3 * Cc);
  if (wv == 0) {  // start Q/K DMA early
    tdm_load_2d(0u,    base + h * 32,      3 * Cc, 128, 32);
    tdm_load_2d(8192u, base + Cc + h * 32, 3 * Cc, 128, 32);
  }
  // V: load rows, store transposed into LDS (overlaps with the TDM)
#pragma unroll
  for (int it = 0; it < 2; ++it) {
    const int chunk = tid + it * 256;
    const int r = chunk >> 2, c8 = (chunk & 3) * 8;
    H8 u = *reinterpret_cast<const H8*>(base + (size_t)r * (3 * Cc) +
                                        2 * Cc + h * 32 + c8);
#pragma unroll
    for (int e = 0; e < 8; ++e) Vts[(c8 + e) * Wn + r] = u.h[e];
  }
  if (wv == 0) __builtin_amdgcn_s_wait_tensorcnt(0);
  __syncthreads();

  const v16h aq = frag_a(&Qs[(wv * 16 + l16) * 32]);
  v8f zero = {};
  v8f s[8];
#pragma unroll
  for (int ni = 0; ni < 8; ++ni) {
    v16h bk = frag_bT(&Ks[(ni * 16 + l16) * 32]);  // B[k][n] = K[n][k]
    s[ni] = wmma_f16(aq, bk, zero);
  }

  const float scale = 0.17677669529663689f;  // 1/sqrt(32)
#pragma unroll
  for (int rr = 0; rr < 8; ++rr) {
    float mx = -3.0e38f;
#pragma unroll
    for (int ni = 0; ni < 8; ++ni) {
      s[ni][rr] *= scale;
      mx = fmaxf(mx, s[ni][rr]);
    }
#pragma unroll
    for (int m = 1; m < 16; m <<= 1) mx = fmaxf(mx, __shfl_xor(mx, m, 32));
    float sum = 0.f;
#pragma unroll
    for (int ni = 0; ni < 8; ++ni) {
      float e = __expf(s[ni][rr] - mx);
      s[ni][rr] = e;
      sum += e;
    }
#pragma unroll
    for (int m = 1; m < 16; m <<= 1) sum += __shfl_xor(sum, m, 32);
    const float inv = 1.f / sum;
    const int prow = wv * 16 + 8 * hf + rr;
#pragma unroll
    for (int ni = 0; ni < 8; ++ni)
      Ps[prow * Wn + ni * 16 + l16] = (_Float16)(s[ni][rr] * inv);
  }
  __syncthreads();

#pragma unroll
  for (int nt = 0; nt < 2; ++nt) {
    v8f acc = zero;
#pragma unroll
    for (int ks = 0; ks < 4; ++ks) {
      v16h ap = frag_a(&Ps[(wv * 16 + l16) * Wn + ks * 32]);
      v16h bv = frag_bT(&Vts[(nt * 16 + l16) * Wn + ks * 32]);
      acc = wmma_f16(ap, bv, acc);
    }
#pragma unroll
    for (int rr = 0; rr < 8; ++rr) {
      obuf[(rowbase + wv * 16 + 8 * hf + rr) * Cc + h * 32 + nt * 16 + l16] =
          (_Float16)acc[rr];
    }
  }
}

// ---------------------------------------------------------------------------
extern "C" void kernel_launch(void* const* d_in, const int* in_sizes, int n_in,
                              void* d_out, int out_size, void* d_ws, size_t ws_size,
                              hipStream_t stream) {
  (void)in_sizes; (void)n_in; (void)out_size; (void)ws_size;
  const float* x     = (const float*)d_in[0];
  const float* ln1_g = (const float*)d_in[1];
  const float* ln1_b = (const float*)d_in[2];
  const float* in_w  = (const float*)d_in[3];
  const float* in_b  = (const float*)d_in[4];
  const float* out_w = (const float*)d_in[5];
  const float* out_b = (const float*)d_in[6];
  const float* ln2_g = (const float*)d_in[7];
  const float* ln2_b = (const float*)d_in[8];
  const float* w1    = (const float*)d_in[9];
  const float* b1    = (const float*)d_in[10];
  const float* w2    = (const float*)d_in[11];
  const float* b2    = (const float*)d_in[12];
  float* out = (float*)d_out;

  char* ws = (char*)d_ws;
  size_t off = 0;
  auto wsAlloc = [&](size_t bytes) -> char* {
    char* p = ws + off;
    off += (bytes + 255) & ~(size_t)255;
    return p;
  };
  // f16 weights, native (N,K) layout
  _Float16* wqkvH = (_Float16*)wsAlloc((size_t)3 * Cc * Cc * 2);  // 768x256
  _Float16* woH   = (_Float16*)wsAlloc((size_t)Cc * Cc * 2);      // 256x256
  _Float16* w1H   = (_Float16*)wsAlloc((size_t)HID * Cc * 2);     // 1024x256
  _Float16* w2H   = (_Float16*)wsAlloc((size_t)Cc * HID * 2);     // 256x1024
  float*    x1    = (float*)wsAlloc((size_t)MTOK * Cc * 4);       // 128 MB
  _Float16* hbuf  = (_Float16*)wsAlloc((size_t)MTOK * Cc * 2);    // 64 MB
  // 256 MB region: qkv(192MB)+obuf(64MB), later reused as MLP hidden (256MB)
  char* region    = wsAlloc((size_t)MTOK * HID * 2);
  _Float16* qkv   = (_Float16*)region;
  _Float16* obuf  = (_Float16*)(region + (size_t)MTOK * 3 * Cc * 2);
  _Float16* mbuf  = (_Float16*)region;

  // 1) weights -> f16 (layout preserved: (out,in) == N-major)
  wcvt_kernel<<<(3 * Cc * Cc + 255) / 256, 256, 0, stream>>>(in_w, wqkvH, 3 * Cc * Cc);
  wcvt_kernel<<<(Cc * Cc + 255) / 256, 256, 0, stream>>>(out_w, woH, Cc * Cc);
  wcvt_kernel<<<(HID * Cc + 255) / 256, 256, 0, stream>>>(w1, w1H, HID * Cc);
  wcvt_kernel<<<(Cc * HID + 255) / 256, 256, 0, stream>>>(w2, w2H, Cc * HID);

  // 2) LN1 + shift + transpose -> f16 token rows (shifted order)
  ln1_shift_kernel<<<MTOK / 16, 256, 0, stream>>>(x, ln1_g, ln1_b, hbuf);

  // 3) QKV projection (131072x256 @ 768x256^T)
  gemm_wmma_kernel<EPI_QKV><<<dim3(MTOK / 128, (3 * Cc) / 128), 256, 32768, stream>>>(
      hbuf, wqkvH, in_b, Cc, 3 * Cc, qkv, nullptr, nullptr);

  // 4) windowed attention, per (b, window, head)
  attn_kernel<<<Bc * NW * Hh, 256, 57344, stream>>>(qkv, obuf);

  // 5) out-proj + un-shift + residual -> x1 fp32 (B,T,C)
  gemm_wmma_kernel<EPI_OUTPROJ><<<dim3(MTOK / 128, Cc / 128), 256, 32768, stream>>>(
      obuf, woH, out_b, Cc, Cc, nullptr, x1, x);

  // 6) LN2 -> f16 (reuse hbuf)
  ln2_kernel<<<MTOK, 256, 0, stream>>>(x1, ln2_g, ln2_b, hbuf);

  // 7) MLP up + GELU (131072x256 @ 1024x256^T) -> mbuf (aliases qkv region)
  gemm_wmma_kernel<EPI_GELU><<<dim3(MTOK / 128, HID / 128), 256, 32768, stream>>>(
      hbuf, w1H, b1, Cc, HID, mbuf, nullptr, nullptr);

  // 8) MLP down + residual + transposed store to (B,C,T)
  gemm_wmma_kernel<EPI_MLP2><<<dim3(MTOK / 128, Cc / 128), 256, 32768, stream>>>(
      mbuf, w2H, b2, HID, Cc, nullptr, out, x1);
}
